// MultiHeadAttentionDynamicGCN_90658169684736
// MI455X (gfx1250) — compile-verified, matrix operand order checked by reading, not run on hardware
//
#include <hip/hip_runtime.h>

// Problem dims (from reference)
#define B_    4
#define T_    64
#define N_    128
#define FIN_  64
#define HID_  128
#define OUT_  256
#define H_    8
#define LAG_  8
#define DD_   8
#define EPS_  1e-8f

typedef __bf16 bf16;
typedef __attribute__((ext_vector_type(16))) __bf16 v16bf;
typedef __attribute__((ext_vector_type(8)))  float  v8f;

// ---------------------------------------------------------------------------
// WMMA helpers (V_WMMA_F32_16X16X32_BF16, wave32)
// A fragment layout (16-bit, 16x32):
//   lanes 0-15 : M=lane, VGPR v<4 -> K = 2v,2v+1 ; v>=4 -> K = 16+2(v-4)..
//   lanes 16-31: same rows, K offset +8 within each half
// B operand supplied as B^T rows (N x K row-major) -> identical load pattern.
// C/D: VGPR r holds (M = r + 8*(lane>=16), N = lane&15).
// ---------------------------------------------------------------------------
__device__ __forceinline__ int frag_kbase(int v, int khalf) {
  return (v < 4) ? (v * 2 + khalf * 8) : (16 + (v - 4) * 2 + khalf * 8);
}

__device__ __forceinline__ v16bf frag_load_rm(const bf16* p16, int ld, int k0) {
  int lane  = threadIdx.x & 31;
  int khalf = lane >> 4;
  const bf16* rp = p16 + (size_t)(lane & 15) * ld + k0;
  v16bf f;
#pragma unroll
  for (int v = 0; v < 8; ++v) {
    int kb = frag_kbase(v, khalf);
    f[2 * v]     = rp[kb];
    f[2 * v + 1] = rp[kb + 1];
  }
  return f;
}

__device__ __forceinline__ v8f wmma_bf16(v16bf a, v16bf b, v8f c) {
  return __builtin_amdgcn_wmma_f32_16x16x32_bf16(false, a, false, b, (short)0, c,
                                                 false, false);
}

// ---------------------------------------------------------------------------
// Kernel 1: X -> bf16 copy, feature-centered Xc (f32), row sum-of-squares SS.
// One wave per (b,t,n) row of 64 features.
// ---------------------------------------------------------------------------
__global__ void __launch_bounds__(256) k_cvt_x(const float* __restrict__ X,
                                               bf16* __restrict__ Xbf,
                                               float* __restrict__ Xc,
                                               float* __restrict__ SS) {
  int wave = threadIdx.x >> 5;
  int lane = threadIdx.x & 31;
  int row  = blockIdx.x * 8 + wave;  // over B*T*N
  const float* xr = X + (size_t)row * FIN_;
  float x0 = xr[lane], x1 = xr[lane + 32];
  float s = x0 + x1;
#pragma unroll
  for (int off = 16; off; off >>= 1) s += __shfl_xor(s, off, 32);
  float mean = s * (1.0f / FIN_);
  float c0 = x0 - mean, c1 = x1 - mean;
  float q = c0 * c0 + c1 * c1;
#pragma unroll
  for (int off = 16; off; off >>= 1) q += __shfl_xor(q, off, 32);
  Xbf[(size_t)row * FIN_ + lane]      = (bf16)x0;
  Xbf[(size_t)row * FIN_ + lane + 32] = (bf16)x1;
  Xc[(size_t)row * FIN_ + lane]       = c0;
  Xc[(size_t)row * FIN_ + lane + 32]  = c1;
  if (lane == 0) SS[row] = q;
}

// ---------------------------------------------------------------------------
// Kernel 2: batched transpose f32 -> bf16  (dst[b][c][r] = src[b][r][c])
// ---------------------------------------------------------------------------
__global__ void k_tr(const float* __restrict__ src, bf16* __restrict__ dst,
                     int batch, int rows, int cols) {
  int total = batch * rows * cols;
  for (int i = blockIdx.x * blockDim.x + threadIdx.x; i < total;
       i += gridDim.x * blockDim.x) {
    int b = i / (rows * cols), rc = i % (rows * cols);
    int r = rc / cols, c = rc % cols;
    dst[(size_t)b * rows * cols + (size_t)c * rows + r] = (bf16)src[i];
  }
}

// ---------------------------------------------------------------------------
// Kernel 3: per-head projection  Out[h] = relu(Xbf @ W[h] + b[h])  (bf16 out)
// ---------------------------------------------------------------------------
__global__ void __launch_bounds__(256) k_qkv(const bf16* __restrict__ Xbf,
                                             const bf16* __restrict__ WT,  // [H][HID][FIN]
                                             const float* __restrict__ bias,
                                             bf16* __restrict__ Out) {     // [H][R][HID]
  int h = blockIdx.z, rt = blockIdx.x;
  int wave = threadIdx.x >> 5, lane = threadIdx.x & 31;
  int c0 = wave * 16;
  const bf16* A16 = Xbf + (size_t)rt * 16 * FIN_;
  const bf16* B16 = WT + ((size_t)h * HID_ + c0) * FIN_;
  v8f acc = {};
#pragma unroll
  for (int k0 = 0; k0 < FIN_; k0 += 32) {
    v16bf a = frag_load_rm(A16, FIN_, k0);
    v16bf b = frag_load_rm(B16, FIN_, k0);
    acc = wmma_bf16(a, b, acc);
  }
  int n = lane & 15, mh = (lane >> 4) * 8;
  float bv = bias[h * HID_ + c0 + n];
  bf16* outp = Out + (size_t)h * (size_t)(B_ * T_ * N_) * HID_ +
               (size_t)rt * 16 * HID_ + c0 + n;
#pragma unroll
  for (int r = 0; r < 8; ++r) {
    float v = acc[r] + bv;
    v = v > 0.f ? v : 0.f;
    outp[(size_t)(mh + r) * HID_] = (bf16)v;
  }
}

// ---------------------------------------------------------------------------
// Kernel 4: fused scores -> softmax -> agg, per (b,t, head, 16-row chunk).
// Scores/weights stay in LDS; agg stages Vcat k-chunks in LDS (coalesced
// b128 global loads instead of per-lane scattered gathers), prefetching the
// next chunk with global_prefetch_b8.
// ---------------------------------------------------------------------------
__global__ void __launch_bounds__(256, 1) k_attn(const bf16* __restrict__ Q,
                                                 const bf16* __restrict__ K,
                                                 const bf16* __restrict__ V,
                                                 bf16* __restrict__ AGG) {
  __shared__ float S[16][LAG_ * N_];   // 64 KB scores (f32)
  __shared__ bf16  Wl[16][LAG_ * N_];  // 32 KB softmaxed weights (bf16)
  __shared__ bf16  Vs[32][HID_];       //  8 KB staged Vcat k-chunk
  __shared__ float red[16][16];

  int bt = blockIdx.x;  // b*T + t
  int h  = blockIdx.y;
  int nt = blockIdx.z;  // 16-row chunk of N
  int t  = bt % T_;
  int b  = bt / T_;
  int wave = threadIdx.x >> 5, lane = threadIdx.x & 31;
  int n = lane & 15, khalf = lane >> 4;

  size_t headoff = ((size_t)h * B_ + b) * T_;
  const bf16* Qb = Q + ((headoff + t) * N_ + (size_t)nt * 16) * HID_;
  const bf16* Kb = K + headoff * N_ * HID_;
  const bf16* Vb = V + headoff * N_ * HID_;

  // ---- hoist the 4 A fragments of the Q chunk (reused by all 8 col tiles)
  v16bf qf[4];
#pragma unroll
  for (int i = 0; i < 4; ++i) qf[i] = frag_load_rm(Qb, HID_, i * 32);

  // ---- scores = Q_chunk @ Kcat^T * 1/sqrt(HID) ----
  const float scale = 0.08838834764831845f;
  for (int ct = wave; ct < 64; ct += 8) {
    int col0 = ct * 16;
    int kr = col0 + n;            // Kcat row for this lane's column
    int l  = kr >> 7, m = kr & (N_ - 1);
    int tl  = t - l;
    int tcl = tl < 0 ? 0 : tl;
    const bf16* krow = Kb + ((size_t)tcl * N_ + m) * HID_;
    v8f acc = {};
#pragma unroll
    for (int i = 0; i < 4; ++i) {
      int k0 = i * 32;
      v16bf bfr;
#pragma unroll
      for (int v = 0; v < 8; ++v) {
        int kb = frag_kbase(v, khalf);
        float e0 = (float)krow[k0 + kb];
        float e1 = (float)krow[k0 + kb + 1];
        bfr[2 * v]     = (bf16)(tl < 0 ? 0.f : e0);
        bfr[2 * v + 1] = (bf16)(tl < 0 ? 0.f : e1);
      }
      acc = wmma_bf16(qf[i], bfr, acc);
    }
    int mh = khalf * 8;
#pragma unroll
    for (int r = 0; r < 8; ++r) S[mh + r][col0 + n] = acc[r] * scale;
  }
  __syncthreads();

  // ---- softmax over 1024 per row (16 threads per row) ----
  {
    int row = threadIdx.x >> 4;
    int sub = threadIdx.x & 15;
    float mx = -3.4e38f;
    for (int c = sub; c < LAG_ * N_; c += 16) mx = fmaxf(mx, S[row][c]);
    red[row][sub] = mx;
    __syncthreads();
    float rmax = red[row][0];
#pragma unroll
    for (int i = 1; i < 16; ++i) rmax = fmaxf(rmax, red[row][i]);
    __syncthreads();
    float sum = 0.f;
    for (int c = sub; c < LAG_ * N_; c += 16) {
      float e = __expf(S[row][c] - rmax);
      S[row][c] = e;
      sum += e;
    }
    red[row][sub] = sum;
    __syncthreads();
    float rsum = 0.f;
#pragma unroll
    for (int i = 0; i < 16; ++i) rsum += red[row][i];
    float inv = 1.0f / rsum;
    for (int c = sub; c < LAG_ * N_; c += 16)
      Wl[row][c] = (bf16)(S[row][c] * inv);
  }

  // ---- agg = w @ Vcat  (K = 1024), V k-chunks staged through LDS ----
  {
    int d0 = wave * 16;  // 8 waves cover HID=128
    // staging map: 8 threads per Vcat row, 16 bf16 (32 B) each
    int srow = threadIdx.x >> 3;          // 0..31
    int scol = (threadIdx.x & 7) * 16;    // 0,16,...,112
    v8f acc = {};
    for (int k0 = 0; k0 < LAG_ * N_; k0 += 32) {
      __syncthreads();  // previous chunk fully consumed (also covers Wl ready)
      {
        int kr = k0 + srow;
        int l = kr >> 7, m = kr & (N_ - 1);
        int tl  = t - l;
        int tcl = tl < 0 ? 0 : tl;
        const uint4* src = (const uint4*)(Vb + ((size_t)tcl * N_ + m) * HID_ + scol);
        uint4 z = {0u, 0u, 0u, 0u};
        uint4 w0 = tl < 0 ? z : src[0];
        uint4 w1 = tl < 0 ? z : src[1];
        uint4* dst = (uint4*)&Vs[srow][scol];
        dst[0] = w0;
        dst[1] = w1;
        // prefetch next chunk's source line
        int kr2 = kr + 32;
        if (kr2 < LAG_ * N_) {
          int l2 = kr2 >> 7, m2 = kr2 & (N_ - 1);
          int tl2 = t - l2;
          int tc2 = tl2 < 0 ? 0 : tl2;
          __builtin_prefetch(Vb + ((size_t)tc2 * N_ + m2) * HID_ + scol, 0, 1);
        }
      }
      __syncthreads();
      v16bf a = frag_load_rm(&Wl[0][0], LAG_ * N_, k0);
      v16bf bfr;
#pragma unroll
      for (int v = 0; v < 8; ++v) {
        int kb = frag_kbase(v, khalf);
        bfr[2 * v]     = Vs[kb][d0 + n];
        bfr[2 * v + 1] = Vs[kb + 1][d0 + n];
      }
      acc = wmma_bf16(a, bfr, acc);
    }
    int mh = khalf * 8;
    bf16* outp = AGG + ((headoff + t) * N_ + (size_t)nt * 16) * HID_ + d0 + n;
#pragma unroll
    for (int r = 0; r < 8; ++r) outp[(size_t)(mh + r) * HID_] = (bf16)acc[r];
  }
}

// ---------------------------------------------------------------------------
// Kernel 5: per-head output proj. ATTN[., h*HID+c] = relu([Q|AGG] @ Wo[h] + bo)
// ---------------------------------------------------------------------------
__global__ void __launch_bounds__(256) k_wo(const bf16* __restrict__ Q,
                                            const bf16* __restrict__ AGG,
                                            const bf16* __restrict__ WoT,  // [H][HID][2*HID]
                                            const float* __restrict__ bo,
                                            bf16* __restrict__ ATTN) {
  int h = blockIdx.z, rt = blockIdx.x;
  int wave = threadIdx.x >> 5, lane = threadIdx.x & 31;
  int c0 = wave * 16;
  size_t hbase = (size_t)h * (size_t)(B_ * T_ * N_) * HID_;
  const bf16* Qh = Q + hbase + (size_t)rt * 16 * HID_;
  const bf16* Gh = AGG + hbase + (size_t)rt * 16 * HID_;
  const bf16* B16 = WoT + ((size_t)h * HID_ + c0) * (2 * HID_);
  v8f acc = {};
#pragma unroll
  for (int k0 = 0; k0 < 2 * HID_; k0 += 32) {
    const bf16* A16 = (k0 < HID_) ? (Qh + k0) : (Gh + (k0 - HID_));
    v16bf a = frag_load_rm(A16, HID_, 0);
    v16bf b = frag_load_rm(B16, 2 * HID_, k0);
    acc = wmma_bf16(a, b, acc);
  }
  int n = lane & 15, mh = (lane >> 4) * 8;
  float bv = bo[h * HID_ + c0 + n];
  bf16* outp = ATTN + (size_t)rt * 16 * (H_ * HID_) + h * HID_ + c0 + n;
#pragma unroll
  for (int r = 0; r < 8; ++r) {
    float v = acc[r] + bv;
    v = v > 0.f ? v : 0.f;
    outp[(size_t)(mh + r) * (H_ * HID_)] = (bf16)v;
  }
}

// ---------------------------------------------------------------------------
// Kernel 6: hg = relu(ATTN @ Wg + bg)   (32768 x 1024) @ (1024 x 256)
// ---------------------------------------------------------------------------
__global__ void __launch_bounds__(256) k_wg(const bf16* __restrict__ ATTN,
                                            const bf16* __restrict__ WgT,  // [OUT][1024]
                                            const float* __restrict__ bg,
                                            bf16* __restrict__ HG) {
  int rt = blockIdx.x;
  int wave = threadIdx.x >> 5, lane = threadIdx.x & 31;
  int c0 = (blockIdx.y * 8 + wave) * 16;
  const bf16* A16 = ATTN + (size_t)rt * 16 * (H_ * HID_);
  const bf16* B16 = WgT + (size_t)c0 * (H_ * HID_);
  v8f acc = {};
  for (int k0 = 0; k0 < H_ * HID_; k0 += 32) {
    v16bf a = frag_load_rm(A16, H_ * HID_, k0);
    v16bf b = frag_load_rm(B16, H_ * HID_, k0);
    acc = wmma_bf16(a, b, acc);
  }
  int n = lane & 15, mh = (lane >> 4) * 8;
  float bv = bg[c0 + n];
  bf16* outp = HG + (size_t)rt * 16 * OUT_ + c0 + n;
#pragma unroll
  for (int r = 0; r < 8; ++r) {
    float v = acc[r] + bv;
    v = v > 0.f ? v : 0.f;
    outp[(size_t)(mh + r) * OUT_] = (bf16)v;
  }
}

// ---------------------------------------------------------------------------
// Kernel 7: delay-correlation adjacency (f32 VALU; 2.3 G FMA, L2-resident)
// ---------------------------------------------------------------------------
__global__ void __launch_bounds__(256) k_adj(const float* __restrict__ Xc,
                                             const float* __restrict__ SS,
                                             const float* __restrict__ alpha,
                                             bf16* __restrict__ ADJ) {
  int tid = blockIdx.x * blockDim.x + threadIdx.x;
  if (tid >= B_ * N_ * N_) return;
  int b = tid >> 14;
  int m = (tid >> 7) & (N_ - 1);
  int nn = tid & (N_ - 1);
  float al[DD_ + 1], amax = -3.4e38f;
#pragma unroll
  for (int d = 0; d <= DD_; ++d) {
    al[d] = alpha[((size_t)m * N_ + nn) * (DD_ + 1) + d];
    amax = fmaxf(amax, al[d]);
  }
  float asum = 0.f;
#pragma unroll
  for (int d = 0; d <= DD_; ++d) {
    al[d] = __expf(al[d] - amax);
    asum += al[d];
  }
  float ainv = 1.f / asum;
  const float4* Xb = (const float4*)(Xc + (size_t)b * T_ * N_ * FIN_);
  const float* SSb = SS + (size_t)b * T_ * N_;
  float adjv = 0.f;
  for (int d = 0; d <= DD_; ++d) {
    float acc = 0.f;
    for (int t = 0; t < T_ - d; ++t) {
      const float4* xi = Xb + ((size_t)t * N_ + m) * (FIN_ / 4);
      const float4* xj = Xb + ((size_t)(t + d) * N_ + nn) * (FIN_ / 4);
      float num = 0.f;
#pragma unroll
      for (int f = 0; f < FIN_ / 4; ++f) {
        float4 a4 = xi[f], b4 = xj[f];
        num += a4.x * b4.x + a4.y * b4.y + a4.z * b4.z + a4.w * b4.w;
      }
      float den = sqrtf(SSb[t * N_ + m] * SSb[(t + d) * N_ + nn]) + EPS_;
      acc += num / den;
    }
    adjv += al[d] * ainv * (acc / (float)(T_ - d));
  }
  ADJ[tid] = (bf16)adjv;
}

// ---------------------------------------------------------------------------
// Kernel 8: out[b,t] = adj[b] (128x128) @ hg[b,t] (128x256), f32 output
// ---------------------------------------------------------------------------
__global__ void __launch_bounds__(256) k_final(const bf16* __restrict__ ADJ,
                                               const bf16* __restrict__ HG,
                                               float* __restrict__ out) {
  int bt = blockIdx.x;
  int b = bt / T_;
  int wave = threadIdx.x >> 5, lane = threadIdx.x & 31;
  int m0 = blockIdx.y * 16;
  int o0 = (blockIdx.z * 8 + wave) * 16;
  int n = lane & 15, khalf = lane >> 4;
  const bf16* A16 = ADJ + (size_t)b * N_ * N_ + (size_t)m0 * N_;
  const bf16* Hb = HG + (size_t)bt * N_ * OUT_;
  v8f acc = {};
#pragma unroll
  for (int k0 = 0; k0 < N_; k0 += 32) {
    v16bf a = frag_load_rm(A16, N_, k0);
    v16bf bfr;  // B operand stored K-major: strided gathers (L2-resident)
#pragma unroll
    for (int v = 0; v < 8; ++v) {
      int kb = frag_kbase(v, khalf);
      bfr[2 * v]     = Hb[(size_t)(k0 + kb) * OUT_ + o0 + n];
      bfr[2 * v + 1] = Hb[(size_t)(k0 + kb + 1) * OUT_ + o0 + n];
    }
    acc = wmma_bf16(a, bfr, acc);
  }
  int mh = khalf * 8;
  float* outp = out + ((size_t)bt * N_ + m0) * OUT_ + o0 + n;
#pragma unroll
  for (int r = 0; r < 8; ++r) outp[(size_t)(mh + r) * OUT_] = acc[r];
}

// ---------------------------------------------------------------------------
extern "C" void kernel_launch(void* const* d_in, const int* in_sizes, int n_in,
                              void* d_out, int out_size, void* d_ws,
                              size_t ws_size, hipStream_t stream) {
  (void)in_sizes; (void)n_in; (void)out_size; (void)ws_size;
  const float* X     = (const float*)d_in[0];
  const float* Wq    = (const float*)d_in[1];
  const float* bq    = (const float*)d_in[2];
  const float* Wk    = (const float*)d_in[3];
  const float* bk    = (const float*)d_in[4];
  const float* Wv    = (const float*)d_in[5];
  const float* bv    = (const float*)d_in[6];
  const float* Wo    = (const float*)d_in[7];
  const float* bo    = (const float*)d_in[8];
  const float* alpha = (const float*)d_in[9];
  const float* Wg    = (const float*)d_in[10];
  const float* bg    = (const float*)d_in[11];
  float* out = (float*)d_out;

  const size_t R = (size_t)B_ * T_ * N_;  // 32768 rows
  char* p = (char*)d_ws;
  auto take = [&](size_t bytes) -> char* {
    char* r = p;
    p += (bytes + 255) & ~(size_t)255;
    return r;
  };
  bf16*  Xbf  = (bf16*) take(R * FIN_ * 2);
  float* Xc   = (float*)take(R * FIN_ * 4);
  float* SS   = (float*)take(R * 4);
  bf16*  WqT  = (bf16*) take((size_t)H_ * HID_ * FIN_ * 2);
  bf16*  WkT  = (bf16*) take((size_t)H_ * HID_ * FIN_ * 2);
  bf16*  WvT  = (bf16*) take((size_t)H_ * HID_ * FIN_ * 2);
  bf16*  WoT  = (bf16*) take((size_t)H_ * HID_ * 2 * HID_ * 2);
  bf16*  WgT  = (bf16*) take((size_t)OUT_ * H_ * HID_ * 2);
  bf16*  Qb   = (bf16*) take((size_t)H_ * R * HID_ * 2);
  bf16*  Kb   = (bf16*) take((size_t)H_ * R * HID_ * 2);
  bf16*  Vb   = (bf16*) take((size_t)H_ * R * HID_ * 2);
  bf16*  AGGb = (bf16*) take((size_t)H_ * R * HID_ * 2);
  bf16*  ATTN = (bf16*) take(R * H_ * HID_ * 2);
  bf16*  HG   = (bf16*) take(R * OUT_ * 2);
  bf16*  ADJ  = (bf16*) take((size_t)B_ * N_ * N_ * 2);

  hipLaunchKernelGGL(k_cvt_x, dim3(R / 8), dim3(256), 0, stream, X, Xbf, Xc, SS);
  hipLaunchKernelGGL(k_tr, dim3(256), dim3(256), 0, stream, Wq, WqT, H_, FIN_, HID_);
  hipLaunchKernelGGL(k_tr, dim3(256), dim3(256), 0, stream, Wk, WkT, H_, FIN_, HID_);
  hipLaunchKernelGGL(k_tr, dim3(256), dim3(256), 0, stream, Wv, WvT, H_, FIN_, HID_);
  hipLaunchKernelGGL(k_tr, dim3(256), dim3(256), 0, stream, Wo, WoT, H_, 2 * HID_, HID_);
  hipLaunchKernelGGL(k_tr, dim3(256), dim3(256), 0, stream, Wg, WgT, 1, H_ * HID_, OUT_);
  hipLaunchKernelGGL(k_qkv, dim3(R / 16, 1, H_), dim3(256), 0, stream, Xbf, WqT, bq, Qb);
  hipLaunchKernelGGL(k_qkv, dim3(R / 16, 1, H_), dim3(256), 0, stream, Xbf, WkT, bk, Kb);
  hipLaunchKernelGGL(k_qkv, dim3(R / 16, 1, H_), dim3(256), 0, stream, Xbf, WvT, bv, Vb);
  hipLaunchKernelGGL(k_attn, dim3(B_ * T_, H_, N_ / 16), dim3(256), 0, stream,
                     Qb, Kb, Vb, AGGb);
  hipLaunchKernelGGL(k_wo, dim3(R / 16, 1, H_), dim3(256), 0, stream, Qb, AGGb,
                     WoT, bo, ATTN);
  hipLaunchKernelGGL(k_wg, dim3(R / 16, 2, 1), dim3(256), 0, stream, ATTN, WgT,
                     bg, HG);
  hipLaunchKernelGGL(k_adj, dim3((B_ * N_ * N_) / 256), dim3(256), 0, stream,
                     Xc, SS, alpha, ADJ);
  hipLaunchKernelGGL(k_final, dim3(B_ * T_, N_ / 16, 2), dim3(256), 0, stream,
                     ADJ, HG, out);
}
